// SpatialSelfCrossAttention_87960930222518
// MI455X (gfx1250) — matrix operand order repaired and checked by reading, not compiled, not run on hardware
//
#include <hip/hip_runtime.h>
#include <hip/hip_bf16.h>

// ---------------------------------------------------------------------------
// Problem constants: B=32, C=256, H=W=32 -> N=1024
// ---------------------------------------------------------------------------
#define BB 32
#define CDIM 256
#define NKEY 1024

typedef _Float16 half8  __attribute__((ext_vector_type(8)));
typedef _Float16 v16h   __attribute__((ext_vector_type(16)));
typedef float    v8f    __attribute__((ext_vector_type(8)));

// ---------------------------------------------------------------------------
// WMMA fragment loaders (wave32, 16x16x32 f16 -> f32)
//
// A (16x32, M x K), 16-bit layout: lane = (m | khalf<<4). Lane holds
//   K = khalf*8 .. +8  (elements 0..7)  and  K = 16+khalf*8 .. +8 (elements 8..15)
// -> two contiguous 16B chunks from a row-major [row][K] array.
//
// B (32x16, K x N), 16-bit layout: lane = (n | khalf<<4). Lane holds column n,
//   K = khalf*8 .. +8 (elements 0..7) and K = 16+khalf*8 .. +8 (elements 8..15)
//   -> two 16B chunks from a [col][K] array.
//
// C/D (16x16 f32): element r of v8f is (m = r + 8*(lane>>4), n = lane&15).
// ---------------------------------------------------------------------------
__device__ __forceinline__ v16h load_frag_a(const _Float16* base, int row0, int k0,
                                            int ld, int lane) {
  int m  = lane & 15;
  int kh = lane >> 4;
  const _Float16* p = base + (size_t)(row0 + m) * ld + k0 + kh * 8;
  half8 lo = *(const half8*)p;
  half8 hi = *(const half8*)(p + 16);
  return __builtin_shufflevector(lo, hi, 0, 1, 2, 3, 4, 5, 6, 7,
                                 8, 9, 10, 11, 12, 13, 14, 15);
}

__device__ __forceinline__ v16h load_frag_b(const _Float16* base, int col0, int k0,
                                            int ld, int lane) {
  int n  = lane & 15;
  int kh = lane >> 4;
  const _Float16* p = base + (size_t)(col0 + n) * ld + k0 + kh * 8;
  half8 lo = *(const half8*)p;
  half8 hi = *(const half8*)(p + 16);
  return __builtin_shufflevector(lo, hi, 0, 1, 2, 3, 4, 5, 6, 7,
                                 8, 9, 10, 11, 12, 13, 14, 15);
}

__device__ __forceinline__ v8f wmma16(v16h a, v16h b, v8f c) {
  return __builtin_amdgcn_wmma_f32_16x16x32_f16(false, a, false, b, (short)0, c,
                                                false, false);
}

// ---------------------------------------------------------------------------
// Kernel 1: convert 5 weight matrices (256x256 f32, row-major [c_out][c]) to f16
// ---------------------------------------------------------------------------
__global__ __launch_bounds__(256) void cvt_weights(const float* wq, const float* wk,
                                                   const float* wv, const float* wps,
                                                   const float* wpc, _Float16* out) {
  int id = blockIdx.x * 256 + threadIdx.x;      // 5 * 65536 total
  const float* srcs[5] = {wq, wk, wv, wps, wpc};
  int wi = id >> 16;
  int e  = id & 65535;
  out[id] = (_Float16)srcs[wi][e];
}

// ---------------------------------------------------------------------------
// Kernel 2: tiled transpose + f32->f16: x[b][c][n] -> xt[b][n][c]
// block (32,8), grid (N/32, C/32, B)
// ---------------------------------------------------------------------------
__global__ __launch_bounds__(256) void transpose_cvt(const float* __restrict__ x,
                                                     _Float16* __restrict__ xt) {
  __shared__ float t[32][33];
  int b  = blockIdx.z;
  int n0 = blockIdx.x * 32;
  int c0 = blockIdx.y * 32;
  const float* xb = x + (size_t)b * CDIM * NKEY;
  _Float16*    xo = xt + (size_t)b * NKEY * CDIM;
  int tx = threadIdx.x, ty = threadIdx.y;
#pragma unroll
  for (int i = 0; i < 4; ++i) {
    int c = c0 + ty + i * 8;
    t[ty + i * 8][tx] = xb[(size_t)c * NKEY + n0 + tx];
  }
  __syncthreads();
#pragma unroll
  for (int i = 0; i < 4; ++i) {
    int n = n0 + ty + i * 8;
    xo[(size_t)n * CDIM + c0 + tx] = (_Float16)t[tx][ty + i * 8];
  }
}

// ---------------------------------------------------------------------------
// Kernel 3: generic NT f16 GEMM with WMMA.
//   Out[i][j] = sum_k A[i][k] * Bt[j][k]  (+ bias[j] or bias[i]), f16 output.
// block (32,4): one wave per 16x16 tile; grid (Ncols/64, Mrows/16, batch).
// ---------------------------------------------------------------------------
__global__ __launch_bounds__(128) void gemm_nt_f16(
    const _Float16* __restrict__ A, const _Float16* __restrict__ Bt,
    const float* __restrict__ bias, _Float16* __restrict__ Out,
    int K, int lda, int ldb, int ldo,
    size_t bsA, size_t bsB, size_t bsO, int biasRow) {
  int lane   = threadIdx.x;
  int tile_n = blockIdx.x * blockDim.y + threadIdx.y;
  int tile_m = blockIdx.y;
  int b      = blockIdx.z;
  const _Float16* Ab = A + bsA * b;
  const _Float16* Bb = Bt + bsB * b;
  _Float16*       Ob = Out + bsO * b;
  int row0 = tile_m * 16, col0 = tile_n * 16;
  v8f acc = {};
  for (int kc = 0; kc < K; kc += 32) {
    v16h af = load_frag_a(Ab, row0, kc, lda, lane);
    v16h bf = load_frag_b(Bb, col0, kc, ldb, lane);
    acc = wmma16(af, bf, acc);
  }
  int n = lane & 15, kh = lane >> 4;
#pragma unroll
  for (int r = 0; r < 8; ++r) {
    int m    = r + 8 * kh;
    float bv = biasRow ? bias[row0 + m] : bias[col0 + n];
    Ob[(size_t)(row0 + m) * ldo + col0 + n] = (_Float16)(acc[r] + bv);
  }
}

// ---------------------------------------------------------------------------
// Kernel 4: fused attention. One workgroup = (batch b, 16-query block qb).
// 256 threads = 8 waves. LDS (160 KB):
//   [0      ..  64K) w1s  : f32 [16][1024]   self logits (kept until branch-1)
//   [64K    .. 128K) w12s : f32 [16][1024]   fused logits; later reused as hs
//   [128K   .. 160K) ps   : f16 [16][1024]   probabilities (reused per branch)
// ---------------------------------------------------------------------------
__device__ __forceinline__ void softmax_row_to_f16(const float* __restrict__ src,
                                                   _Float16* __restrict__ dst,
                                                   int m, int lane) {
  float buf[32];
  float mx = -1e30f;
#pragma unroll
  for (int i = 0; i < 32; ++i) {
    float v = src[m * NKEY + lane + i * 32];
    buf[i]  = v;
    mx      = fmaxf(mx, v);
  }
#pragma unroll
  for (int o = 16; o; o >>= 1) mx = fmaxf(mx, __shfl_xor(mx, o, 32));
  float s = 0.f;
#pragma unroll
  for (int i = 0; i < 32; ++i) {
    float e = __expf(buf[i] - mx);
    buf[i]  = e;
    s += e;
  }
#pragma unroll
  for (int o = 16; o; o >>= 1) s += __shfl_xor(s, o, 32);
  float inv = 1.0f / s;
#pragma unroll
  for (int i = 0; i < 32; ++i)
    dst[m * NKEY + lane + i * 32] = (_Float16)(buf[i] * inv);
}

__global__ __launch_bounds__(256) void attn_fused(
    const _Float16* __restrict__ Qt1, const _Float16* __restrict__ Kt1,
    const _Float16* __restrict__ Qt2, const _Float16* __restrict__ Kt2,
    const _Float16* __restrict__ V,
    const _Float16* __restrict__ Wps16, const _Float16* __restrict__ Wpc16,
    const float* __restrict__ bps, const float* __restrict__ bpc,
    const float* __restrict__ x1,
    float* __restrict__ out1, float* __restrict__ out2) {
  __shared__ __align__(32) char smem[160 * 1024];
  float*    w1s  = (float*)smem;                    // [16][1024] f32
  float*    w12s = (float*)(smem + 64 * 1024);      // [16][1024] f32
  _Float16* ps   = (_Float16*)(smem + 128 * 1024);  // [16][1024] f16
  _Float16* hs   = (_Float16*)(smem + 64 * 1024);   // [16][256]  f16 (overlays w12s)

  int tid  = threadIdx.x;
  int lane = tid & 31;
  int w    = tid >> 5;          // wave 0..7
  int qb   = blockIdx.x;        // query block 0..63
  int b    = blockIdx.y;        // batch
  int q0   = qb * 16;

  size_t boff = (size_t)b * NKEY * CDIM;
  const _Float16* Qt1b = Qt1 + boff;
  const _Float16* Kt1b = Kt1 + boff;
  const _Float16* Qt2b = Qt2 + boff;
  const _Float16* Kt2b = Kt2 + boff;
  const _Float16* Vb   = V + boff;

  // ---- Stage 1: logits w1 = q1^T k1 * s,  w12 = w1 + q2^T k2 * s -------------
  v16h qa1[8], qa2[8];
#pragma unroll
  for (int kc = 0; kc < 8; ++kc) {
    qa1[kc] = load_frag_a(Qt1b, q0, kc * 32, CDIM, lane);
    qa2[kc] = load_frag_a(Qt2b, q0, kc * 32, CDIM, lane);
  }
  int dn = lane & 15, dkh = lane >> 4;
  for (int t = 0; t < 8; ++t) {
    int col0 = (w * 8 + t) * 16;               // key tile
    v8f acc1 = {}, acc2 = {};
#pragma unroll
    for (int kc = 0; kc < 8; ++kc) {
      v16h kb1 = load_frag_b(Kt1b, col0, kc * 32, CDIM, lane);
      acc1 = wmma16(qa1[kc], kb1, acc1);
      v16h kb2 = load_frag_b(Kt2b, col0, kc * 32, CDIM, lane);
      acc2 = wmma16(qa2[kc], kb2, acc2);
    }
#pragma unroll
    for (int r = 0; r < 8; ++r) {
      int m    = r + 8 * dkh;
      float l1 = acc1[r] * 0.0625f;            // 1/sqrt(256)
      float l2 = acc2[r] * 0.0625f;
      w1s[m * NKEY + col0 + dn]  = l1;
      w12s[m * NKEY + col0 + dn] = l1 + l2;
    }
  }
  __syncthreads();

  // ---- Stage 2: softmax(w1+w2) -> ps (cross branch) --------------------------
  softmax_row_to_f16(w12s, ps, 2 * w, lane);
  softmax_row_to_f16(w12s, ps, 2 * w + 1, lane);
  __syncthreads();

  // ---- Stage 3: h12[k][c] = sum_n p[k][n] * v[c][n]  -> hs (f16) -------------
#pragma unroll
  for (int u = 0; u < 2; ++u) {
    int c0 = (w * 2 + u) * 16;
    v8f acc = {};
    for (int nt = 0; nt < 32; ++nt) {
      v16h af = load_frag_a(ps, 0, nt * 32, NKEY, lane);
      v16h bf = load_frag_b(Vb, c0, nt * 32, NKEY, lane);
      acc = wmma16(af, bf, acc);
    }
#pragma unroll
    for (int r = 0; r < 8; ++r)
      hs[(r + 8 * dkh) * CDIM + c0 + dn] = (_Float16)acc[r];
  }
  __syncthreads();

  // ---- Stage 4: out2 = x1 + Wpc*h12 + bpc ; Stage 5: softmax(w1) -> ps -------
#pragma unroll
  for (int u = 0; u < 2; ++u) {
    int c0 = (w * 2 + u) * 16;
    v8f acc = {};
#pragma unroll
    for (int kc = 0; kc < 8; ++kc) {
      v16h af = load_frag_a(hs, 0, kc * 32, CDIM, lane);
      v16h bf = load_frag_b(Wpc16, c0, kc * 32, CDIM, lane);
      acc = wmma16(af, bf, acc);
    }
#pragma unroll
    for (int r = 0; r < 8; ++r) {
      int m = r + 8 * dkh;
      int c = c0 + dn;
      size_t oi = (size_t)b * CDIM * NKEY + (size_t)c * NKEY + q0 + m;
      out2[oi] = acc[r] + bpc[c] + x1[oi];
    }
  }
  softmax_row_to_f16(w1s, ps, 2 * w, lane);      // self branch (w1s untouched)
  softmax_row_to_f16(w1s, ps, 2 * w + 1, lane);
  __syncthreads();

  // ---- Stage 6: h1 -> hs ------------------------------------------------------
#pragma unroll
  for (int u = 0; u < 2; ++u) {
    int c0 = (w * 2 + u) * 16;
    v8f acc = {};
    for (int nt = 0; nt < 32; ++nt) {
      v16h af = load_frag_a(ps, 0, nt * 32, NKEY, lane);
      v16h bf = load_frag_b(Vb, c0, nt * 32, NKEY, lane);
      acc = wmma16(af, bf, acc);
    }
#pragma unroll
    for (int r = 0; r < 8; ++r)
      hs[(r + 8 * dkh) * CDIM + c0 + dn] = (_Float16)acc[r];
  }
  __syncthreads();

  // ---- Stage 7: out1 = x1 + Wps*h1 + bps -------------------------------------
#pragma unroll
  for (int u = 0; u < 2; ++u) {
    int c0 = (w * 2 + u) * 16;
    v8f acc = {};
#pragma unroll
    for (int kc = 0; kc < 8; ++kc) {
      v16h af = load_frag_a(hs, 0, kc * 32, CDIM, lane);
      v16h bf = load_frag_b(Wps16, c0, kc * 32, CDIM, lane);
      acc = wmma16(af, bf, acc);
    }
#pragma unroll
    for (int r = 0; r < 8; ++r) {
      int m = r + 8 * dkh;
      int c = c0 + dn;
      size_t oi = (size_t)b * CDIM * NKEY + (size_t)c * NKEY + q0 + m;
      out1[oi] = acc[r] + bps[c] + x1[oi];
    }
  }
}

// ---------------------------------------------------------------------------
// Host-side launch
// ---------------------------------------------------------------------------
extern "C" void kernel_launch(void* const* d_in, const int* in_sizes, int n_in,
                              void* d_out, int out_size, void* d_ws, size_t ws_size,
                              hipStream_t stream) {
  (void)in_sizes; (void)n_in; (void)out_size; (void)ws_size;

  const float* x1  = (const float*)d_in[0];
  const float* x2  = (const float*)d_in[1];
  const float* Wq  = (const float*)d_in[2];
  const float* bq  = (const float*)d_in[3];
  const float* Wk  = (const float*)d_in[4];
  const float* bk  = (const float*)d_in[5];
  const float* Wv  = (const float*)d_in[6];
  const float* bv  = (const float*)d_in[7];
  const float* Wps = (const float*)d_in[8];
  const float* bps = (const float*)d_in[9];
  const float* Wpc = (const float*)d_in[10];
  const float* bpc = (const float*)d_in[11];

  float* out1 = (float*)d_out;
  float* out2 = out1 + (size_t)BB * CDIM * NKEY;

  // workspace carve (f16): 7 tensors of [32][1024][256] + 5 weight matrices
  const size_t T = (size_t)BB * NKEY * CDIM;   // 8,388,608 halves / tensor
  _Float16* ws   = (_Float16*)d_ws;
  _Float16* Xt1  = ws;          // [b][n][c]
  _Float16* Xt2  = ws + 1 * T;
  _Float16* Qt1  = ws + 2 * T;  // [b][n][c]
  _Float16* Kt1  = ws + 3 * T;  // [b][n][c]  (k transposed)
  _Float16* Qt2  = ws + 4 * T;
  _Float16* Kt2  = ws + 5 * T;
  _Float16* Vm   = ws + 6 * T;  // [b][c][n]
  _Float16* W16  = ws + 7 * T;  // 5 x 65536
  _Float16* Wq16 = W16;
  _Float16* Wk16 = W16 + 1 * 65536;
  _Float16* Wv16 = W16 + 2 * 65536;
  _Float16* Wps16 = W16 + 3 * 65536;
  _Float16* Wpc16 = W16 + 4 * 65536;

  // 1. weights -> f16
  cvt_weights<<<5 * 65536 / 256, 256, 0, stream>>>(Wq, Wk, Wv, Wps, Wpc, W16);

  // 2. x -> xt (f16, [n][c])
  transpose_cvt<<<dim3(NKEY / 32, CDIM / 32, BB), dim3(32, 8), 0, stream>>>(x1, Xt1);
  transpose_cvt<<<dim3(NKEY / 32, CDIM / 32, BB), dim3(32, 8), 0, stream>>>(x2, Xt2);

  // 3. QKV projections (all WMMA GEMMs)
  const size_t NC = (size_t)NKEY * CDIM;
  dim3 gblk(32, 4);
  // Qt1[n][co] = sum_c Xt1[n][c] Wq[co][c] + bq[co]
  gemm_nt_f16<<<dim3(CDIM / 64, NKEY / 16, BB), gblk, 0, stream>>>(
      Xt1, Wq16, bq, Qt1, CDIM, CDIM, CDIM, CDIM, NC, 0, NC, 0);
  gemm_nt_f16<<<dim3(CDIM / 64, NKEY / 16, BB), gblk, 0, stream>>>(
      Xt1, Wk16, bk, Kt1, CDIM, CDIM, CDIM, CDIM, NC, 0, NC, 0);
  gemm_nt_f16<<<dim3(CDIM / 64, NKEY / 16, BB), gblk, 0, stream>>>(
      Xt2, Wq16, bq, Qt2, CDIM, CDIM, CDIM, CDIM, NC, 0, NC, 0);
  gemm_nt_f16<<<dim3(CDIM / 64, NKEY / 16, BB), gblk, 0, stream>>>(
      Xt2, Wk16, bk, Kt2, CDIM, CDIM, CDIM, CDIM, NC, 0, NC, 0);
  // V[co][n] = sum_c Wv[co][c] Xt1[n][c] + bv[co]
  gemm_nt_f16<<<dim3(NKEY / 64, CDIM / 16, BB), gblk, 0, stream>>>(
      Wv16, Xt1, bv, Vm, CDIM, CDIM, CDIM, NKEY, 0, NC, NC, 1);

  // 4. fused attention + output convs + residual
  attn_fused<<<dim3(NKEY / 16, BB), 256, 0, stream>>>(
      Qt1, Kt1, Qt2, Kt2, Vm, Wps16, Wpc16, bps, bpc, x1, out1, out2);
}
